// HadamardCode_63024350101737
// MI455X (gfx1250) — matrix operand order repaired and checked by reading, not compiled, not run on hardware
//
#include <hip/hip_runtime.h>
#include <stdint.h>

typedef __attribute__((ext_vector_type(16))) _Float16 v16h;
typedef __attribute__((ext_vector_type(8)))  float    v8f;
typedef __attribute__((ext_vector_type(4)))  unsigned int v4u;

#define BATCH 512
#define INF   4096      // K of GEMM1
#define OUTF  1023      // code_length - 1
#define NPAD  1024      // padded N / K of GEMM2
#define NCLS  1000
#define DIST  1023.0f

// LDS tile: 128 rows x 32 k-values (f16) = 64B data + 16B pad per row.
#define T_STRIDE 80
#define T_TILE   (128 * T_STRIDE)    // 10240 B
// smem layout: [A0][A1][B0][B1], each T_TILE.
#define SMEM_BYTES (4 * T_TILE)      // 40960 B

union Frag { struct { v4u lo, hi; } p; v16h h; };

// ---- CDNA5 async copy: global -> LDS, 16B per lane, tracked by ASYNCcnt ----
__device__ __forceinline__ void async_b128(uint32_t lds_off, const void* g) {
  asm volatile("global_load_async_to_lds_b128 %0, %1, off"
               :: "v"(lds_off), "v"(g) : "memory");
}
__device__ __forceinline__ void wait_async_partial() {   // current stage done, next in flight
  asm volatile("s_wait_asynccnt 4" ::: "memory");
}
__device__ __forceinline__ void wait_async_all() {
  asm volatile("s_wait_asynccnt 0" ::: "memory");
}

// Explicit LDS (addrspace 3) 16B read from a 32-bit LDS byte offset -> ds_load_b128.
__device__ __forceinline__ v4u lds_read_b128(uint32_t off) {
  const __attribute__((address_space(3))) v4u* p =
      (const __attribute__((address_space(3))) v4u*)(uint32_t)off;
  return *p;
}

// A fragment (16x32 f16). ISA: lanes 0-15 -> M=lane, K={0..7,16..23}; lanes 16-31 -> K+8.
// Per lane: 16B at row + 16*sel, and 16B at row + 32 + 16*sel.
__device__ __forceinline__ v16h lds_a_frag(uint32_t tile, int mLocal, int lane) {
  uint32_t row = tile + (uint32_t)(mLocal + (lane & 15)) * T_STRIDE + ((lane >> 4) << 4);
  Frag f;
  f.p.lo = lds_read_b128(row);
  f.p.hi = lds_read_b128(row + 32);
  return f.h;
}
// B fragment (32x16 f16), tile stored [n][32k]. Lanes 0-15 -> N=lane, K={0..7,16..23};
// lanes 16-31 -> K+8 (same as A).  Per lane: 16B at row + 16*sel, and 16B at row + 32 + 16*sel.
__device__ __forceinline__ v16h lds_b_frag(uint32_t tile, int nLocal, int lane) {
  uint32_t row = tile + (uint32_t)(nLocal + (lane & 15)) * T_STRIDE + ((lane >> 4) << 4);
  Frag f;
  f.p.lo = lds_read_b128(row);
  f.p.hi = lds_read_b128(row + 32);
  return f.h;
}

// Double-buffered async-LDS WMMA mainloop. A: f16 [M][lda] row-major, B: f16 [N][ldb]
// row-major (operand transposed via B-fragment layout). WG tile 128x128, 8 waves (32x64 each).
__device__ __forceinline__ void wmma_mainloop(const _Float16* __restrict__ Ag, int lda,
                                              const _Float16* __restrict__ Bg, int ldb,
                                              int K, int mWG, int nWG,
                                              uint32_t smemBase, v8f acc[2][4]) {
  const int tid  = threadIdx.x;
  const int lane = tid & 31;
  const int wave = tid >> 5;
  const int wm = wave & 3, wn = wave >> 2;

  auto issue = [&](int it, int buf) {
    int kb = it * 32;
#pragma unroll
    for (int h = 0; h < 2; ++h) {                    // A: 512 16B chunks / 256 threads
      int c = tid + h * 256;
      int row = c >> 2, seg = c & 3;                 // 4 x 16B per 32-k row
      const void* g = Ag + (size_t)(mWG + row) * lda + kb + seg * 8;
      async_b128(smemBase + (uint32_t)buf * T_TILE + row * T_STRIDE + seg * 16, g);
    }
#pragma unroll
    for (int h = 0; h < 2; ++h) {                    // B: 512 16B chunks / 256 threads
      int c = tid + h * 256;
      int row = c >> 2, seg = c & 3;
      const void* g = Bg + (size_t)(nWG + row) * ldb + kb + seg * 8;
      async_b128(smemBase + (uint32_t)(2 + buf) * T_TILE + row * T_STRIDE + seg * 16, g);
    }
  };

  const int nit = K / 32;
  issue(0, 0);
  for (int it = 0; it < nit; ++it) {
    int cur = it & 1;
    if (it + 1 < nit) { issue(it + 1, cur ^ 1); wait_async_partial(); }
    else              { wait_async_all(); }
    __syncthreads();                                  // all waves' copies visible
    uint32_t aT = smemBase + (uint32_t)cur * T_TILE;
    uint32_t bT = smemBase + (uint32_t)(2 + cur) * T_TILE;
    v16h a[2], bf[4];
#pragma unroll
    for (int i = 0; i < 2; ++i) a[i]  = lds_a_frag(aT, wm * 32 + 16 * i, lane);
#pragma unroll
    for (int j = 0; j < 4; ++j) bf[j] = lds_b_frag(bT, wn * 64 + 16 * j, lane);
#pragma unroll
    for (int i = 0; i < 2; ++i)
#pragma unroll
      for (int j = 0; j < 4; ++j)
        acc[i][j] = __builtin_amdgcn_wmma_f32_16x16x32_f16(
            false, a[i], false, bf[j], (short)0, acc[i][j], false, false);
    __syncthreads();                                  // reads done before buffer reuse
  }
}

// ---------------- conversion kernels ----------------
__global__ void cvt_x(const float* __restrict__ x, _Float16* __restrict__ Xh, int n) {
  int i = blockIdx.x * blockDim.x + threadIdx.x;
  if (i < n) Xh[i] = (_Float16)x[i];
}
// Wh[n][k], n padded to 1024 (zeros), f16.
__global__ void cvt_wh(const float* __restrict__ W, _Float16* __restrict__ Wh) {
  int i = blockIdx.x * blockDim.x + threadIdx.x;
  if (i >= NPAD * INF) return;
  int n = i >> 12;                  // / 4096
  int k = i & (INF - 1);
  float w = (n < OUTF) ? W[(size_t)n * INF + k] : 0.f;
  Wh[i] = (_Float16)w;
}
// Lh[c][j], both padded to 1024 (zeros), f16.
__global__ void cvt_lh(const float* __restrict__ L, _Float16* __restrict__ Lh) {
  int i = blockIdx.x * blockDim.x + threadIdx.x;
  if (i >= NPAD * NPAD) return;
  int c = i >> 10;
  int j = i & (NPAD - 1);
  float l = (c < NCLS && j < OUTF) ? L[(size_t)c * OUTF + j] : 0.f;
  Lh[i] = (_Float16)l;
}

// ---------------- GEMM1: H = tanh(X @ W^T + b) ----------------
__global__ __launch_bounds__(256) void gemm1_tanh(const _Float16* __restrict__ Xh,
                                                  const _Float16* __restrict__ Wh,
                                                  const float* __restrict__ bias,
                                                  _Float16* __restrict__ H) {
  __shared__ __align__(16) unsigned char smem[SMEM_BYTES];
  uint32_t smemBase = (uint32_t)(uintptr_t)&smem[0];
  v8f acc[2][4];
#pragma unroll
  for (int i = 0; i < 2; ++i)
#pragma unroll
    for (int j = 0; j < 4; ++j) acc[i][j] = v8f{};

  int mWG = blockIdx.y * 128, nWG = blockIdx.x * 128;
  wmma_mainloop(Xh, INF, Wh, INF, INF, mWG, nWG, smemBase, acc);

  int lane = threadIdx.x & 31;
  int wave = threadIdx.x >> 5;
  int wm = wave & 3, wn = wave >> 2;
  int mBase = mWG + wm * 32, nBase = nWG + wn * 64;
  int n0 = lane & 15, rsel = (lane >> 4) << 3;
#pragma unroll
  for (int i = 0; i < 2; ++i)
#pragma unroll
    for (int j = 0; j < 4; ++j) {
      int n = nBase + 16 * j + n0;
      float bn = (n < OUTF) ? bias[n] : 0.0f;
#pragma unroll
      for (int r = 0; r < 8; ++r) {
        int m = mBase + 16 * i + r + rsel;
        float hv = (n < OUTF) ? tanhf(acc[i][j][r] + bn) : 0.0f;
        H[(size_t)m * NPAD + n] = (_Float16)hv;
      }
    }
}

// ---------------- GEMM2: inv = 1/max(1023 - H@L^T, eps)^p ----------------
__global__ __launch_bounds__(256) void gemm2_dist(const _Float16* __restrict__ H,
                                                  const _Float16* __restrict__ Lh,
                                                  const float* __restrict__ epsp,
                                                  const float* __restrict__ powp,
                                                  float* __restrict__ out) {
  __shared__ __align__(16) unsigned char smem[SMEM_BYTES];
  uint32_t smemBase = (uint32_t)(uintptr_t)&smem[0];
  float eps = epsp[0], power = powp[0];
  v8f acc[2][4];
#pragma unroll
  for (int i = 0; i < 2; ++i)
#pragma unroll
    for (int j = 0; j < 4; ++j) acc[i][j] = v8f{};

  int mWG = blockIdx.y * 128, nWG = blockIdx.x * 128;
  wmma_mainloop(H, NPAD, Lh, NPAD, NPAD, mWG, nWG, smemBase, acc);

  int lane = threadIdx.x & 31;
  int wave = threadIdx.x >> 5;
  int wm = wave & 3, wn = wave >> 2;
  int mBase = mWG + wm * 32, nBase = nWG + wn * 64;
  int n0 = lane & 15, rsel = (lane >> 4) << 3;
#pragma unroll
  for (int i = 0; i < 2; ++i)
#pragma unroll
    for (int j = 0; j < 4; ++j) {
      int n = nBase + 16 * j + n0;
      if (n < NCLS) {
#pragma unroll
        for (int r = 0; r < 8; ++r) {
          int m = mBase + 16 * i + r + rsel;
          float d = DIST - acc[i][j][r];       // sum |h - l| via labels in {+-1}, |h|<=1
          d = fmaxf(d, eps);
          float inv = (power == 1.0f) ? (1.0f / d) : (1.0f / powf(d, power));
          out[(size_t)m * NCLS + n] = inv;
        }
      }
    }
}

// ---------------- row normalization ----------------
__global__ void rownorm(float* __restrict__ out) {
  __shared__ float sdata[256];
  int r = blockIdx.x;
  float s = 0.f;
  for (int c = threadIdx.x; c < NCLS; c += 256) s += out[(size_t)r * NCLS + c];
  sdata[threadIdx.x] = s;
  __syncthreads();
  for (int st = 128; st > 0; st >>= 1) {
    if ((int)threadIdx.x < st) sdata[threadIdx.x] += sdata[threadIdx.x + st];
    __syncthreads();
  }
  float inv = 1.0f / sdata[0];
  for (int c = threadIdx.x; c < NCLS; c += 256) out[(size_t)r * NCLS + c] *= inv;
}

extern "C" void kernel_launch(void* const* d_in, const int* in_sizes, int n_in,
                              void* d_out, int out_size, void* d_ws, size_t ws_size,
                              hipStream_t stream) {
  const float* x      = (const float*)d_in[0];
  const float* W      = (const float*)d_in[1];
  const float* b      = (const float*)d_in[2];
  const float* labels = (const float*)d_in[3];
  const float* eps    = (const float*)d_in[4];
  const float* power  = (const float*)d_in[5];
  (void)in_sizes; (void)n_in; (void)out_size; (void)ws_size;

  char* ws = (char*)d_ws;
  _Float16* Xh = (_Float16*)(ws);                   // 512*4096*2  = 4 MiB
  _Float16* Wh = (_Float16*)(ws + (4u  << 20));     // 1024*4096*2 = 8 MiB
  _Float16* Lh = (_Float16*)(ws + (12u << 20));     // 1024*1024*2 = 2 MiB
  _Float16* H  = (_Float16*)(ws + (14u << 20));     // 512*1024*2  = 1 MiB
  float* out   = (float*)d_out;

  cvt_x <<<(BATCH * INF + 255) / 256, 256, 0, stream>>>(x, Xh, BATCH * INF);
  cvt_wh<<<(NPAD * INF) / 256, 256, 0, stream>>>(W, Wh);
  cvt_lh<<<(NPAD * NPAD) / 256, 256, 0, stream>>>(labels, Lh);
  gemm1_tanh<<<dim3(8, 4), 256, 0, stream>>>(Xh, Wh, b, H);
  gemm2_dist<<<dim3(8, 4), 256, 0, stream>>>(H, Lh, eps, power, out);
  rownorm<<<BATCH, 256, 0, stream>>>(out);
}
